// GPT_62929860821180
// MI455X (gfx1250) — matrix operand order repaired and checked by reading, not compiled, not run on hardware
//
#include <hip/hip_runtime.h>

// ---- model constants (match reference) ----
#define B_  2
#define S_  2048
#define D_  1024
#define H_  8
#define HD_ 128
#define V_  32000
#define L_  6
#define M_  (B_ * S_)        // 4096 token rows
#define DFF (4 * D_)         // 4096
#define SCALE_F 0.08838834764831845f   // 128^-0.5

typedef __attribute__((ext_vector_type(16))) _Float16 v16h;
typedef __attribute__((ext_vector_type(8)))  _Float16 v8h;
typedef __attribute__((ext_vector_type(8)))  float    v8f;
typedef __attribute__((ext_vector_type(4)))  unsigned int u32x4;
typedef __attribute__((ext_vector_type(8)))  int i32x8;
typedef __attribute__((ext_vector_type(4)))  int i32x4;

static __device__ __forceinline__ v16h cat8(v8h lo, v8h hi) {
  return __builtin_shufflevector(lo, hi, 0,1,2,3,4,5,6,7,8,9,10,11,12,13,14,15);
}
static __device__ __forceinline__ v8f wmma16(v16h a, v16h b, v8f c) {
  return __builtin_amdgcn_wmma_f32_16x16x32_f16(false, a, false, b, (short)0, c,
                                                false, false);
}
static __device__ __forceinline__ v8f vzero8() {
  v8f z;
#pragma unroll
  for (int i = 0; i < 8; ++i) z[i] = 0.f;
  return z;
}
static __device__ __forceinline__ unsigned lds_off_of(const void* p) {
  return (unsigned)(unsigned long long)p;   // low 32 bits of flat LDS addr = LDS offset
}

// TDM: DMA a tile1 x tile0 tile (f16, dim0 contiguous) from a [dim1, dim0]
// row-major tensor into LDS.  gaddr = byte address of tile origin.
static __device__ __forceinline__ void tdm_load_tile_f16(
    unsigned lds_addr, unsigned long long gaddr, int dim0, int dim1,
    int tile0, int tile1) {
  u32x4 g0;
  g0[0] = 1u;                                            // count=1, user mode
  g0[1] = lds_addr;                                      // LDS byte address
  g0[2] = (unsigned)(gaddr & 0xffffffffu);               // global addr [31:0]
  g0[3] = (unsigned)((gaddr >> 32) & 0x01ffffffu) | (2u << 30);  // [56:32] | type=2
  i32x8 g1;
  g1[0] = 1 << 16;                                       // data_size = 2 bytes
  g1[1] = (dim0 & 0xffff) << 16;                         // tensor_dim0 [15:0]
  g1[2] = ((unsigned)dim0 >> 16) | ((dim1 & 0xffff) << 16);  // dim0[31:16] | dim1[15:0]
  g1[3] = ((unsigned)dim1 >> 16) | (tile0 << 16);        // dim1[31:16] | tile_dim0
  g1[4] = tile1 & 0xffff;                                // tile_dim1 (tile_dim2=0)
  g1[5] = dim0;                                          // tensor_dim0_stride [31:0]
  g1[6] = 0;                                             // stride hi / dim1_stride lo
  g1[7] = 0;
  i32x4 z4;
  z4[0] = z4[1] = z4[2] = z4[3] = 0;
#if defined(__clang_major__) && __clang_major__ >= 23
  i32x8 z8;
#pragma unroll
  for (int i = 0; i < 8; ++i) z8[i] = 0;
  __builtin_amdgcn_tensor_load_to_lds(g0, g1, z4, z4, z8, 0);
#else
  __builtin_amdgcn_tensor_load_to_lds(g0, g1, z4, z4, 0);
#endif
}

// ---------------- f32 -> f16 transposing weight conversion ----------------
// in: [K,N] row-major f32;  out: [N,K] row-major f16
__global__ __launch_bounds__(256) void cvt_t_f32_f16_kernel(
    const float* __restrict__ in, _Float16* __restrict__ out, int K, int N) {
  long i = (long)blockIdx.x * 256 + threadIdx.x;
  if (i < (long)K * N) {
    long k = i / N, n = i % N;
    out[n * K + k] = (_Float16)in[i];
  }
}

// ---------------- embedding gather ----------------
__global__ __launch_bounds__(256) void embed_kernel(
    const int* __restrict__ x, const float* __restrict__ emb,
    float* __restrict__ h) {
  long i = (long)blockIdx.x * 256 + threadIdx.x;
  if (i < (long)M_ * D_) {
    long t = i / D_, d = i % D_;
    h[i] = emb[(long)x[t] * D_ + d];
  }
}

// ---------------- rel-pos bias table: bt[h][delta], pre-scaled ----------------
__global__ __launch_bounds__(256) void relbias_kernel(
    const float* __restrict__ rel_emb, float* __restrict__ bt) {
  int i = blockIdx.x * 256 + threadIdx.x;
  if (i >= H_ * S_) return;
  int hh = i / S_, d = i % S_;   // d = i - j >= 0
  int bucket;
  if (d < 16) {
    bucket = d;
  } else {
    float nf = (float)d;
    int vl = 16 + (int)(__logf(nf * (1.f / 16.f)) * (16.f / 2.0794415416798357f));
    bucket = vl < 31 ? vl : 31;
  }
  bt[(long)hh * S_ + d] = rel_emb[bucket * H_ + hh] * SCALE_F;
}

// ---------------- LayerNorm (one row per block) ----------------
template <int OUT_HALF>
__global__ __launch_bounds__(256) void layernorm_kernel(
    const float* __restrict__ x, const float* __restrict__ gw,
    const float* __restrict__ bw, void* __restrict__ outp) {
  __shared__ float red[256];
  const int row = blockIdx.x, tid = threadIdx.x;
  const float* xr = x + (long)row * D_;
  float s = 0.f;
  for (int i = tid; i < D_; i += 256) s += xr[i];
  red[tid] = s; __syncthreads();
  for (int off = 128; off > 0; off >>= 1) {
    if (tid < off) red[tid] += red[tid + off];
    __syncthreads();
  }
  const float mean = red[0] * (1.f / D_);
  __syncthreads();
  float vs = 0.f;
  for (int i = tid; i < D_; i += 256) { float d = xr[i] - mean; vs += d * d; }
  red[tid] = vs; __syncthreads();
  for (int off = 128; off > 0; off >>= 1) {
    if (tid < off) red[tid] += red[tid + off];
    __syncthreads();
  }
  const float rstd = rsqrtf(red[0] * (1.f / D_) + 1e-5f);
  for (int i = tid; i < D_; i += 256) {
    float y = (xr[i] - mean) * rstd * gw[i] + bw[i];
    if (OUT_HALF) ((_Float16*)outp)[(long)row * D_ + i] = (_Float16)y;
    else          ((float*)outp)[(long)row * D_ + i] = y;
  }
}

// ------- WMMA GEMM: C = A[M,K](f16) * BT[N,K](f16)^T + bias, TDM-staged -------
// EPI: 0 = f32 out + bias        (lm head)
//      1 = f32 out + bias + res  (residual adds)
//      2 = f16 out + bias        (q/k)
//      3 = f16 out + bias + relu (ffn mid)
//      4 = f16 out + bias, TRANSPOSED store out[N,M] (v -> V^T)
template <int EPI>
__global__ __launch_bounds__(256) void gemm_f16_wmma(
    const _Float16* __restrict__ A, const _Float16* __restrict__ BT,
    const float* __restrict__ bias, const float* __restrict__ res,
    void* __restrict__ outp, int M, int N, int K) {
  __shared__ __align__(64) _Float16 As[2][128][32];   // [m][k]
  __shared__ __align__(64) _Float16 Bs[2][128][32];   // [n][k]
  const int tid = threadIdx.x;
  const int lane = tid & 31, wid = tid >> 5;
  const int wm = wid >> 1, wn = wid & 1;              // 4x2 wave grid
  const int c = lane & 15, g = lane >> 4;
  const int m0 = blockIdx.y * 128, n0 = blockIdx.x * 128;

  v8f acc[2][4];
#pragma unroll
  for (int i = 0; i < 2; ++i)
#pragma unroll
    for (int j = 0; j < 4; ++j) acc[i][j] = vzero8();

  const unsigned long long abase = (unsigned long long)A;
  const unsigned long long bbase = (unsigned long long)BT;

  // prologue: DMA first K-tile into buffer 0 (wave 0 drives the TDM)
  if (wid == 0) {
    tdm_load_tile_f16(lds_off_of(&As[0][0][0]),
                      abase + ((unsigned long long)m0 * K) * 2, K, M, 32, 128);
    tdm_load_tile_f16(lds_off_of(&Bs[0][0][0]),
                      bbase + ((unsigned long long)n0 * K) * 2, K, N, 32, 128);
  }

  int buf = 0;
  for (int k0 = 0; k0 < K; k0 += 32) {
    if (wid == 0) {
      if (k0 + 32 < K) {   // issue next tile, then wait only for current pair
        tdm_load_tile_f16(lds_off_of(&As[buf ^ 1][0][0]),
                          abase + ((unsigned long long)m0 * K + k0 + 32) * 2,
                          K, M, 32, 128);
        tdm_load_tile_f16(lds_off_of(&Bs[buf ^ 1][0][0]),
                          bbase + ((unsigned long long)n0 * K + k0 + 32) * 2,
                          K, N, 32, 128);
        __builtin_amdgcn_s_wait_tensorcnt(2);
      } else {
        __builtin_amdgcn_s_wait_tensorcnt(0);
      }
    }
    __syncthreads();   // current tile visible to all waves

    v16h af[2], bf[4];
#pragma unroll
    for (int mi = 0; mi < 2; ++mi) {
      const _Float16* p = &As[buf][wm * 32 + mi * 16 + c][g * 16];
      af[mi] = cat8(*(const v8h*)p, *(const v8h*)(p + 8));
    }
#pragma unroll
    for (int ni = 0; ni < 4; ++ni)
      bf[ni] = *(const v16h*)(&Bs[buf][wn * 64 + ni * 16 + c][g * 16]);

#pragma unroll
    for (int mi = 0; mi < 2; ++mi)
#pragma unroll
      for (int ni = 0; ni < 4; ++ni)
        acc[mi][ni] = wmma16(af[mi], bf[ni], acc[mi][ni]);

    __syncthreads();   // reads done before next DMA overwrites this buffer
    buf ^= 1;
  }

#pragma unroll
  for (int mi = 0; mi < 2; ++mi)
#pragma unroll
    for (int ni = 0; ni < 4; ++ni) {
      const int col = n0 + wn * 64 + ni * 16 + c;
      const float bv = bias[col];
#pragma unroll
      for (int v = 0; v < 8; ++v) {
        const long row = m0 + wm * 32 + mi * 16 + v + 8 * g;
        float x = acc[mi][ni][v] + bv;
        if (EPI == 1) x += res[row * N + col];
        if (EPI == 3) x = x > 0.f ? x : 0.f;
        if (EPI == 0 || EPI == 1) ((float*)outp)[row * N + col] = x;
        else if (EPI == 4) ((_Float16*)outp)[(long)col * M + row] = (_Float16)x;
        else ((_Float16*)outp)[row * N + col] = (_Float16)x;
      }
    }
}

// ---------------- flash attention: 1 wave = 16 query rows of one (b,h) -------
// q,k: [B*S, H*HD] f16 ; vt: [H*HD, B*S] f16 (transposed V) ; o: [B*S, H*HD]
__global__ __launch_bounds__(128) void attention_kernel(
    const _Float16* __restrict__ q, const _Float16* __restrict__ k,
    const _Float16* __restrict__ vt, const float* __restrict__ bt,
    _Float16* __restrict__ o) {
  __shared__ __align__(64) _Float16 pbuf[4][16][32];   // per-wave P bounce
  const int lane = threadIdx.x & 31, wv = threadIdx.x >> 5;
  const int c = lane & 15, g = lane >> 4;
  const int i0 = (blockIdx.x * 4 + wv) * 16;
  const int hh = blockIdx.y, b = blockIdx.z;
  const long rs = H_ * HD_;                             // 1024
  const long base = (long)b * S_ * rs + hh * HD_;

  // Q fragments (16 rows x 128 hd -> 4 A-frags)
  v16h qa[4];
  {
    const _Float16* qr = q + base + (long)(i0 + c) * rs;
#pragma unroll
    for (int ks = 0; ks < 4; ++ks) {
      const _Float16* p = qr + ks * 32 + g * 16;
      qa[ks] = cat8(*(const v8h*)p, *(const v8h*)(p + 8));
    }
  }

  float mrow[8], lrow[8];
  v8f Oa[8];
#pragma unroll
  for (int i = 0; i < 8; ++i) { mrow[i] = -1e30f; lrow[i] = 0.f; Oa[i] = vzero8(); }
  const float* btr = bt + (long)hh * S_;
  const _Float16* vtb = vt + (long)(hh * HD_ + c) * M_ + (long)b * S_;

  for (int j0 = 0; j0 <= i0; j0 += 32) {
    // ---- hoist all K fragments for this 32-key step (clause-friendly) ----
    v16h kb[2][4];
#pragma unroll
    for (int t = 0; t < 2; ++t) {
      const _Float16* kr = k + base + (long)(j0 + t * 16 + c) * rs;
#pragma unroll
      for (int ks = 0; ks < 4; ++ks)
        kb[t][ks] = *(const v16h*)(kr + ks * 32 + g * 16);
    }
    // ---- S = Q K^T ----
    v8f st[2];
#pragma unroll
    for (int t = 0; t < 2; ++t) {
      v8f s = vzero8();
#pragma unroll
      for (int ks = 0; ks < 4; ++ks) s = wmma16(qa[ks], kb[t][ks], s);
#pragma unroll
      for (int vv = 0; vv < 8; ++vv) {
        const int irow = i0 + vv + 8 * g;
        const int jcol = j0 + t * 16 + c;
        const int d = irow - jcol;
        const float bias = btr[d > 0 ? d : 0];
        const float val = (s[vv] + bias) * SCALE_F;   // bias already carries SCALE
        s[vv] = (jcol <= irow) ? val : -1e30f;
      }
      st[t] = s;
    }
    // ---- online softmax update ----
    float alpha[8];
#pragma unroll
    for (int vv = 0; vv < 8; ++vv) {
      float x = fmaxf(st[0][vv], st[1][vv]);
#pragma unroll
      for (int msk = 8; msk >= 1; msk >>= 1) x = fmaxf(x, __shfl_xor(x, msk, 32));
      const float nm = fmaxf(mrow[vv], x);
      alpha[vv] = __expf(mrow[vv] - nm);
      mrow[vv] = nm;
      const float p0 = __expf(st[0][vv] - nm);
      const float p1 = __expf(st[1][vv] - nm);
      st[0][vv] = p0; st[1][vv] = p1;
      float sum = p0 + p1;
#pragma unroll
      for (int msk = 8; msk >= 1; msk >>= 1) sum += __shfl_xor(sum, msk, 32);
      lrow[vv] = lrow[vv] * alpha[vv] + sum;
    }
#pragma unroll
    for (int n = 0; n < 8; ++n)
#pragma unroll
      for (int vv = 0; vv < 8; ++vv) Oa[n][vv] *= alpha[vv];

    // ---- P -> LDS (C-layout to A-layout relayout) ----
#pragma unroll
    for (int vv = 0; vv < 8; ++vv) {
      pbuf[wv][vv + 8 * g][c]      = (_Float16)st[0][vv];
      pbuf[wv][vv + 8 * g][16 + c] = (_Float16)st[1][vv];
    }
    const _Float16* pr = &pbuf[wv][c][0];
    v16h pa = cat8(*(const v8h*)(pr + g * 16), *(const v8h*)(pr + g * 16 + 8));

    // ---- hoist all V^T fragments (contiguous b128 loads), then O += P V ----
    v16h vb[8];
#pragma unroll
    for (int n = 0; n < 8; ++n)
      vb[n] = *(const v16h*)(vtb + (long)(n * 16) * M_ + j0 + g * 16);
#pragma unroll
    for (int n = 0; n < 8; ++n) Oa[n] = wmma16(pa, vb[n], Oa[n]);
  }

  // ---- normalize + store to [B,S,H*HD] f16 ----
#pragma unroll
  for (int n = 0; n < 8; ++n)
#pragma unroll
    for (int vv = 0; vv < 8; ++vv) {
      const long row = (long)b * S_ + i0 + vv + 8 * g;
      o[row * rs + hh * HD_ + n * 16 + c] = (_Float16)(Oa[n][vv] / lrow[vv]);
    }
}

// =======================  host side  =======================
extern "C" void kernel_launch(void* const* d_in, const int* in_sizes, int n_in,
                              void* d_out, int out_size, void* d_ws, size_t ws_size,
                              hipStream_t stream) {
  (void)in_sizes; (void)n_in; (void)out_size; (void)ws_size;
  const int*   x       = (const int*)d_in[0];
  const float* tok_emb = (const float*)d_in[1];
  const float* rel_emb = (const float*)d_in[2];
  const float* ln_g    = (const float*)d_in[3];
  const float* ln_b    = (const float*)d_in[4];
  const float* Wqkv    = (const float*)d_in[5];
  const float* bqkv    = (const float*)d_in[6];
  const float* Wo      = (const float*)d_in[7];
  const float* bo      = (const float*)d_in[8];
  const float* W1      = (const float*)d_in[9];
  const float* b1      = (const float*)d_in[10];
  const float* W2      = (const float*)d_in[11];
  const float* b2      = (const float*)d_in[12];
  const float* lnf_g   = (const float*)d_in[13];
  const float* lnf_b   = (const float*)d_in[14];
  const float* Whead   = (const float*)d_in[15];
  const float* bhead   = (const float*)d_in[16];
  float* out = (float*)d_out;

  // ---- workspace layout ----
  char* ws = (char*)d_ws;
  size_t off = 0;
  auto take = [&](size_t bytes) -> void* {
    void* p = ws + off;
    off += (bytes + 255) & ~(size_t)255;
    return p;
  };
  const long WQKV_E  = (long)L_ * 3 * D_ * (H_ * HD_);
  const long WO_E    = (long)L_ * (H_ * HD_) * D_;
  const long W1_E    = (long)L_ * D_ * DFF;
  const long W2_E    = (long)L_ * DFF * D_;
  const long WHEAD_E = (long)D_ * V_;

  _Float16* WqkvT  = (_Float16*)take(WQKV_E  * 2);   // per-slice [N,K]
  _Float16* WoT    = (_Float16*)take(WO_E    * 2);
  _Float16* W1T    = (_Float16*)take(W1_E    * 2);
  _Float16* W2T    = (_Float16*)take(W2_E    * 2);
  _Float16* WheadT = (_Float16*)take(WHEAD_E * 2);
  float*    h      = (float*)   take((long)M_ * D_ * 4);
  float*    tmp    = (float*)   take((long)M_ * D_ * 4);
  _Float16* xln    = (_Float16*)take((long)M_ * D_ * 2);
  _Float16* qh     = (_Float16*)take((long)M_ * D_ * 2);
  _Float16* kh     = (_Float16*)take((long)M_ * D_ * 2);
  _Float16* vtb    = (_Float16*)take((long)D_ * M_ * 2);  // V^T
  _Float16* ao     = (_Float16*)take((long)M_ * D_ * 2);
  _Float16* mid    = (_Float16*)take((long)M_ * DFF * 2);
  float*    bt     = (float*)   take((long)H_ * S_ * 4);

  auto cvt_t = [&](const float* src, _Float16* dst, int K, int N) {
    long n = (long)K * N;
    cvt_t_f32_f16_kernel<<<dim3((unsigned)((n + 255) / 256)), dim3(256), 0, stream>>>(
        src, dst, K, N);
  };
  for (int l = 0; l < L_; ++l) {
    for (int s3 = 0; s3 < 3; ++s3)
      cvt_t(Wqkv + ((long)l * 3 + s3) * D_ * (H_ * HD_),
            WqkvT + ((long)l * 3 + s3) * D_ * (H_ * HD_), D_, H_ * HD_);
    cvt_t(Wo + (long)l * (H_ * HD_) * D_, WoT + (long)l * (H_ * HD_) * D_,
          H_ * HD_, D_);
    cvt_t(W1 + (long)l * D_ * DFF, W1T + (long)l * D_ * DFF, D_, DFF);
    cvt_t(W2 + (long)l * DFF * D_, W2T + (long)l * DFF * D_, DFF, D_);
  }
  cvt_t(Whead, WheadT, D_, V_);

  relbias_kernel<<<dim3((H_ * S_ + 255) / 256), dim3(256), 0, stream>>>(rel_emb, bt);
  embed_kernel<<<dim3((unsigned)(((long)M_ * D_ + 255) / 256)), dim3(256), 0, stream>>>(
      x, tok_emb, h);

  auto ln_h = [&](const float* src, const float* g_, const float* b_, _Float16* dst) {
    layernorm_kernel<1><<<dim3(M_), dim3(256), 0, stream>>>(src, g_, b_, dst);
  };
  auto ln_f = [&](const float* src, const float* g_, const float* b_, float* dst) {
    layernorm_kernel<0><<<dim3(M_), dim3(256), 0, stream>>>(src, g_, b_, dst);
  };

  const dim3 blk(256);
  auto gemm = [&](int EPI, const _Float16* A, const _Float16* Bm, const float* bias,
                  const float* res, void* outp, int N, int K) {
    dim3 grid(N / 128, M_ / 128);
    switch (EPI) {
      case 0: gemm_f16_wmma<0><<<grid, blk, 0, stream>>>(A, Bm, bias, res, outp, M_, N, K); break;
      case 1: gemm_f16_wmma<1><<<grid, blk, 0, stream>>>(A, Bm, bias, res, outp, M_, N, K); break;
      case 2: gemm_f16_wmma<2><<<grid, blk, 0, stream>>>(A, Bm, bias, res, outp, M_, N, K); break;
      case 4: gemm_f16_wmma<4><<<grid, blk, 0, stream>>>(A, Bm, bias, res, outp, M_, N, K); break;
      default:gemm_f16_wmma<3><<<grid, blk, 0, stream>>>(A, Bm, bias, res, outp, M_, N, K); break;
    }
  };

  for (int l = 0; l < L_; ++l) {
    const float* lg = ln_g + (long)l * 3 * D_;
    const float* lb = ln_b + (long)l * 3 * D_;

    // attn: LN -> q,k,v -> flash attention -> out-proj (+residual)
    ln_h(h, lg + 0 * D_, lb + 0 * D_, xln);
    gemm(2, xln, WqkvT + ((long)l * 3 + 0) * D_ * (H_ * HD_),
         bqkv + ((long)l * 3 + 0) * (H_ * HD_), nullptr, qh, H_ * HD_, D_);
    gemm(2, xln, WqkvT + ((long)l * 3 + 1) * D_ * (H_ * HD_),
         bqkv + ((long)l * 3 + 1) * (H_ * HD_), nullptr, kh, H_ * HD_, D_);
    gemm(4, xln, WqkvT + ((long)l * 3 + 2) * D_ * (H_ * HD_),
         bqkv + ((long)l * 3 + 2) * (H_ * HD_), nullptr, vtb, H_ * HD_, D_);
    attention_kernel<<<dim3(S_ / 64, H_, B_), dim3(128), 0, stream>>>(qh, kh, vtb, bt, ao);
    gemm(1, ao, WoT + (long)l * (H_ * HD_) * D_, bo + (long)l * D_, h, h, D_, H_ * HD_);

    // ffn: LN(LN(h)) -> relu(xW1+b1) -> xW2+b2 (+residual)
    ln_f(h, lg + 1 * D_, lb + 1 * D_, tmp);
    ln_h(tmp, lg + 2 * D_, lb + 2 * D_, xln);
    gemm(3, xln, W1T + (long)l * D_ * DFF, b1 + (long)l * DFF, nullptr, mid, DFF, D_);
    gemm(1, mid, W2T + (long)l * DFF * D_, b2 + (long)l * D_, h, h, D_, DFF);
  }

  // final LN + LM head
  ln_h(h, lnf_g, lnf_b, xln);
  gemm(0, xln, WheadT, bhead, nullptr, out, V_, D_);
}
